// GAT_3350074490930
// MI455X (gfx1250) — matrix operand order repaired and checked by reading, not compiled, not run on hardware
//
#include <hip/hip_runtime.h>
#include <hip/hip_bf16.h>
#include <math.h>

#define NNODES 50000
#define NEDGES 800000
#define ETOT   (NEDGES + NNODES)   // self-loops appended
#define INCH   128
#define HC     128                 // heads * channels, layer 1
#define HEADS  8
#define CH     16
#define OUTC   40
#define NSLOPE 0.2f

typedef __attribute__((ext_vector_type(16))) _Float16 v16h;
typedef __attribute__((ext_vector_type(8)))  float    v8f;

// ---- float <-> order-preserving unsigned key (for atomicMax on floats) ----
__device__ __forceinline__ unsigned enc_f(float f) {
  unsigned u = __float_as_uint(f);
  return (u & 0x80000000u) ? ~u : (u | 0x80000000u);
}
__device__ __forceinline__ float dec_f(unsigned u) {
  unsigned b = (u & 0x80000000u) ? (u & 0x7FFFFFFFu) : ~u;
  return __uint_as_float(b);
}

__device__ __forceinline__ void edge_sd(const int* __restrict__ ei, int e, int& s, int& d) {
  if (e < NEDGES) { s = ei[e]; d = ei[NEDGES + e]; }
  else            { s = e - NEDGES; d = s; }      // self loop
}

// ================= GEMM 1: h1 = X[50000x128] @ W1[128x128] (f16 WMMA, f32 acc)
__global__ void gemm1_wmma(const float* __restrict__ X, const float* __restrict__ W,
                           float* __restrict__ Hout) {
  int wave = threadIdx.x >> 5;
  int lane = threadIdx.x & 31;
  int tile = blockIdx.x * 8 + wave;     // 3125*8 = 25000 tiles exactly
  int mt = tile >> 3;                   // 0..3124
  int nt = tile & 7;                    // 0..7 (== head index)
  int row0 = mt * 16;
  int col0 = nt * 16;
  int hs   = lane >> 4;                 // half select
  int lm   = lane & 15;

  v8f acc = {};
  for (int k0 = 0; k0 < INCH; k0 += 32) {
    // A tile 16x32: lane holds row M=lm; K = (hs?16:0)+0..7 and (hs?24:8)+0..7
    const float* arow = X + (size_t)(row0 + lm) * INCH + k0;
    int ka = hs ? 16 : 0, kb = hs ? 24 : 8;
    v16h a;
#pragma unroll
    for (int i = 0; i < 8; i++) { a[i]   = (_Float16)arow[ka + i];
                                  a[8+i] = (_Float16)arow[kb + i]; }
    // B tile 32x16: lane holds col n=lm; K = k0 + (hs?16:0) + i
    const float* bcol = W + (size_t)(k0 + (hs ? 16 : 0)) * HC + (col0 + lm);
    v16h b;
#pragma unroll
    for (int i = 0; i < 16; i++) b[i] = (_Float16)bcol[(size_t)i * HC];
    acc = __builtin_amdgcn_wmma_f32_16x16x32_f16(false, a, false, b, (short)0, acc,
                                                 false, false);
  }
  // D: acc[v] = D[v + 8*hs][lm]
#pragma unroll
  for (int v = 0; v < 8; v++) {
    int r = row0 + v + (hs ? 8 : 0);
    Hout[(size_t)r * HC + col0 + lm] = acc[v];
  }
}

// ================= GEMM 2: h2 = o1[50000x128] @ W2[128x40] (pad N->48)
__global__ void gemm2_wmma(const float* __restrict__ A, const float* __restrict__ W,
                           float* __restrict__ Hout) {
  int wave = threadIdx.x >> 5;
  int lane = threadIdx.x & 31;
  int tile = blockIdx.x * 8 + wave;
  if (tile >= 3125 * 3) return;
  int mt = tile / 3;
  int nt = tile % 3;
  int row0 = mt * 16;
  int col0 = nt * 16;
  int hs = lane >> 4;
  int lm = lane & 15;
  int col = col0 + lm;

  v8f acc = {};
  for (int k0 = 0; k0 < HC; k0 += 32) {
    const float* arow = A + (size_t)(row0 + lm) * HC + k0;
    int ka = hs ? 16 : 0, kb = hs ? 24 : 8;
    v16h a;
#pragma unroll
    for (int i = 0; i < 8; i++) { a[i]   = (_Float16)arow[ka + i];
                                  a[8+i] = (_Float16)arow[kb + i]; }
    const float* bcol = W + (size_t)(k0 + (hs ? 16 : 0)) * OUTC + col;
    v16h b;
#pragma unroll
    for (int i = 0; i < 16; i++)
      b[i] = (col < OUTC) ? (_Float16)bcol[(size_t)i * OUTC] : (_Float16)0.0f;
    acc = __builtin_amdgcn_wmma_f32_16x16x32_f16(false, a, false, b, (short)0, acc,
                                                 false, false);
  }
#pragma unroll
  for (int v = 0; v < 8; v++) {
    int r = row0 + v + (hs ? 8 : 0);
    if (col < OUTC) Hout[(size_t)r * OUTC + col] = acc[v];
  }
}

// ================= attention coefficients =================
__global__ void alphas1_kernel(const float* __restrict__ h1,
                               const float* __restrict__ aws, const float* __restrict__ awd,
                               float* __restrict__ as, float* __restrict__ ad) {
  int t = blockIdx.x * blockDim.x + threadIdx.x;
  if (t >= NNODES * HEADS) return;
  int n = t >> 3, hh = t & 7;
  const float* hp = h1 + (size_t)n * HC + hh * CH;
  const float* ws = aws + hh * CH;
  const float* wd = awd + hh * CH;
  float ss = 0.f, sd = 0.f;
#pragma unroll
  for (int c = 0; c < CH; c++) { float hv = hp[c]; ss += hv * ws[c]; sd += hv * wd[c]; }
  as[t] = ss; ad[t] = sd;
}

__global__ void alphas2_kernel(const float* __restrict__ h2,
                               const float* __restrict__ aws, const float* __restrict__ awd,
                               float* __restrict__ as, float* __restrict__ ad) {
  int n = blockIdx.x * blockDim.x + threadIdx.x;
  if (n >= NNODES) return;
  const float* hp = h2 + (size_t)n * OUTC;
  float ss = 0.f, sd = 0.f;
#pragma unroll
  for (int c = 0; c < OUTC; c++) { float hv = hp[c]; ss += hv * aws[c]; sd += hv * awd[c]; }
  as[n] = ss; ad[n] = sd;
}

// ================= layer 1 edge phase (8 heads) =================
__global__ void edge_max1(const int* __restrict__ ei, const float* __restrict__ as,
                          const float* __restrict__ ad, float* __restrict__ ebuf,
                          unsigned* __restrict__ mkey) {
  int t = blockIdx.x * blockDim.x + threadIdx.x;
  if (t >= ETOT * HEADS) return;
  int e = t >> 3, hh = t & 7;
  int s, d; edge_sd(ei, e, s, d);
  float v = as[s * HEADS + hh] + ad[d * HEADS + hh];
  v = (v >= 0.f) ? v : NSLOPE * v;
  ebuf[t] = v;
  atomicMax(&mkey[d * HEADS + hh], enc_f(v));
}

__global__ void edge_sum1(const int* __restrict__ ei, float* __restrict__ ebuf,
                          const unsigned* __restrict__ mkey, float* __restrict__ den) {
  int t = blockIdx.x * blockDim.x + threadIdx.x;
  if (t >= ETOT * HEADS) return;
  int e = t >> 3, hh = t & 7;
  int s, d; edge_sd(ei, e, s, d);
  float m = dec_f(mkey[d * HEADS + hh]);
  float ex = expf(ebuf[t] - m);
  ebuf[t] = ex;                               // overwrite with exp
  atomicAdd(&den[d * HEADS + hh], ex);
}

__global__ void edge_agg1(const int* __restrict__ ei, const float* __restrict__ h1,
                          const float* __restrict__ ebuf, const float* __restrict__ den,
                          float* __restrict__ agg) {
  int e = blockIdx.x;                         // one block (128 threads) per edge
  int s, d; edge_sd(ei, e, s, d);
  __shared__ float alpha[HEADS];
  if (threadIdx.x < HEADS)
    alpha[threadIdx.x] = ebuf[(size_t)e * HEADS + threadIdx.x] /
                         (den[d * HEADS + threadIdx.x] + 1e-16f);
  __syncthreads();
  int c = threadIdx.x;                        // 0..127
  int hh = c >> 4;
  atomicAdd(&agg[(size_t)d * HC + c], h1[(size_t)s * HC + c] * alpha[hh]);
}

__global__ void bias1_kernel(float* __restrict__ o1, const float* __restrict__ b1) {
  int t = blockIdx.x * blockDim.x + threadIdx.x;
  if (t >= NNODES * HC) return;
  o1[t] += b1[t & (HC - 1)];
}

// ================= layer 2 edge phase (1 head) =================
__global__ void edge_max2(const int* __restrict__ ei, const float* __restrict__ as,
                          const float* __restrict__ ad, float* __restrict__ ebuf,
                          unsigned* __restrict__ mkey) {
  int e = blockIdx.x * blockDim.x + threadIdx.x;
  if (e >= ETOT) return;
  int s, d; edge_sd(ei, e, s, d);
  float v = as[s] + ad[d];
  v = (v >= 0.f) ? v : NSLOPE * v;
  ebuf[e] = v;
  atomicMax(&mkey[d], enc_f(v));
}

__global__ void edge_sum2(const int* __restrict__ ei, float* __restrict__ ebuf,
                          const unsigned* __restrict__ mkey, float* __restrict__ den) {
  int e = blockIdx.x * blockDim.x + threadIdx.x;
  if (e >= ETOT) return;
  int s, d; edge_sd(ei, e, s, d);
  float ex = expf(ebuf[e] - dec_f(mkey[d]));
  ebuf[e] = ex;
  atomicAdd(&den[d], ex);
}

__global__ void edge_agg2(const int* __restrict__ ei, const float* __restrict__ h2,
                          const float* __restrict__ ebuf, const float* __restrict__ den,
                          float* __restrict__ agg) {
  int e = blockIdx.x;
  int s, d; edge_sd(ei, e, s, d);
  int c = threadIdx.x;                        // 64 threads, 40 active
  if (c >= OUTC) return;
  float alpha = ebuf[e] / (den[d] + 1e-16f);
  atomicAdd(&agg[(size_t)d * OUTC + c], h2[(size_t)s * OUTC + c] * alpha);
}

// ================= final: +b2, log_softmax =================
__global__ void final_logsoftmax(const float* __restrict__ agg, const float* __restrict__ b2,
                                 float* __restrict__ out) {
  int n = blockIdx.x * blockDim.x + threadIdx.x;
  if (n >= NNODES) return;
  float v[OUTC];
  float mx = -INFINITY;
#pragma unroll
  for (int c = 0; c < OUTC; c++) { v[c] = agg[(size_t)n * OUTC + c] + b2[c];
                                   mx = fmaxf(mx, v[c]); }
  float s = 0.f;
#pragma unroll
  for (int c = 0; c < OUTC; c++) s += expf(v[c] - mx);
  float ls = logf(s);
#pragma unroll
  for (int c = 0; c < OUTC; c++) out[(size_t)n * OUTC + c] = v[c] - mx - ls;
}

extern "C" void kernel_launch(void* const* d_in, const int* in_sizes, int n_in,
                              void* d_out, int out_size, void* d_ws, size_t ws_size,
                              hipStream_t stream) {
  const float* x    = (const float*)d_in[0];
  const int*   ei   = (const int*)  d_in[1];
  const float* W1   = (const float*)d_in[2];
  const float* aS1  = (const float*)d_in[3];
  const float* aD1  = (const float*)d_in[4];
  const float* b1   = (const float*)d_in[5];
  const float* W2   = (const float*)d_in[6];
  const float* aS2  = (const float*)d_in[7];
  const float* aD2  = (const float*)d_in[8];
  const float* b2   = (const float*)d_in[9];
  float* out = (float*)d_out;

  char* w = (char*)d_ws;
  float*    h1   = (float*)w;    w += (size_t)NNODES * HC * 4;
  float*    as1  = (float*)w;    w += (size_t)NNODES * HEADS * 4;
  float*    ad1  = (float*)w;    w += (size_t)NNODES * HEADS * 4;
  unsigned* m1   = (unsigned*)w; w += (size_t)NNODES * HEADS * 4;
  float*    den1 = (float*)w;    w += (size_t)NNODES * HEADS * 4;
  float*    e1   = (float*)w;    w += (size_t)ETOT * HEADS * 4;
  float*    o1   = (float*)w;    w += (size_t)NNODES * HC * 4;
  float*    h2   = (float*)w;    w += (size_t)NNODES * OUTC * 4;
  float*    as2  = (float*)w;    w += (size_t)NNODES * 4;
  float*    ad2  = (float*)w;    w += (size_t)NNODES * 4;
  unsigned* m2   = (unsigned*)w; w += (size_t)NNODES * 4;
  float*    den2 = (float*)w;    w += (size_t)NNODES * 4;
  float*    e2   = (float*)w;    w += (size_t)ETOT * 4;
  float*    agg2 = (float*)w;    w += (size_t)NNODES * OUTC * 4;

  // zero-init accumulators; ordered-uint key 0 == -inf for the float atomic max
  hipMemsetAsync(m1,   0, (size_t)NNODES * HEADS * 4, stream);
  hipMemsetAsync(den1, 0, (size_t)NNODES * HEADS * 4, stream);
  hipMemsetAsync(o1,   0, (size_t)NNODES * HC * 4,    stream);
  hipMemsetAsync(m2,   0, (size_t)NNODES * 4,         stream);
  hipMemsetAsync(den2, 0, (size_t)NNODES * 4,         stream);
  hipMemsetAsync(agg2, 0, (size_t)NNODES * OUTC * 4,  stream);

  // ---- layer 1 ----
  gemm1_wmma<<<3125, 256, 0, stream>>>(x, W1, h1);
  alphas1_kernel<<<(NNODES * HEADS + 255) / 256, 256, 0, stream>>>(h1, aS1, aD1, as1, ad1);
  edge_max1<<<(ETOT * HEADS + 255) / 256, 256, 0, stream>>>(ei, as1, ad1, e1, m1);
  edge_sum1<<<(ETOT * HEADS + 255) / 256, 256, 0, stream>>>(ei, e1, m1, den1);
  edge_agg1<<<ETOT, 128, 0, stream>>>(ei, h1, e1, den1, o1);
  bias1_kernel<<<(NNODES * HC + 255) / 256, 256, 0, stream>>>(o1, b1);

  // ---- layer 2 ----
  gemm2_wmma<<<(3125 * 3 + 7) / 8, 256, 0, stream>>>(o1, W2, h2);
  alphas2_kernel<<<(NNODES + 255) / 256, 256, 0, stream>>>(h2, aS2, aD2, as2, ad2);
  edge_max2<<<(ETOT + 255) / 256, 256, 0, stream>>>(ei, as2, ad2, e2, m2);
  edge_sum2<<<(ETOT + 255) / 256, 256, 0, stream>>>(ei, e2, m2, den2);
  edge_agg2<<<ETOT, 64, 0, stream>>>(ei, h2, e2, den2, agg2);

  final_logsoftmax<<<(NNODES + 255) / 256, 256, 0, stream>>>(agg2, b2, out);
}
